// sgnn_10977936408928
// MI455X (gfx1250) — matrix operand order repaired and checked by reading, not compiled, not run on hardware
//
#include <hip/hip_runtime.h>
#include <math.h>

#define B_        64
#define NPEPC     50
#define NPROC     676
#define NV        760          // valid nodes per graph
#define NP        768          // padded nodes (48 * 16)
#define D_        512
#define H_        1024
#define BNC       0.9999950000374997f   // 1/sqrt(1+1e-5)
#define ROWS_PEP  3200         // 64*50
#define ROWS_TOT  46464        // 64*(50+676)
#define MB_NP     24           // NP/32  (32-row M blocks)
#define NB_NP     12           // NP/64  (64-col N blocks)
#define NB_D      8            // D/64

typedef __attribute__((ext_vector_type(16))) __bf16 v16bf;
typedef __attribute__((ext_vector_type(8)))  __bf16 v8bf;
typedef __attribute__((ext_vector_type(8)))  float  v8f;
typedef unsigned short u16;

// ---------- bf16 helpers (bit-exact RNE, buffers typed as u16) ----------
__device__ __forceinline__ u16 f2bf(float f) {
  unsigned u = __float_as_uint(f);
  u += 0x7fffu + ((u >> 16) & 1u);
  return (u16)(u >> 16);
}
__device__ __forceinline__ float bf2f(u16 h) {
  return __uint_as_float(((unsigned)h) << 16);
}

// ---------- WMMA fragment loaders (CDNA5 16x16x32 bf16 layouts) ----------
// A: lane L holds row M = L&15; contiguous K [k0+g*16, +16), g = L>>4.
__device__ __forceinline__ v16bf load_a_frag(const u16* A, int lda, int row, int k0, int lane) {
  int g = lane >> 4;
  return *reinterpret_cast<const v16bf*>(A + (size_t)row * lda + k0 + g * 16);
}
// B: lane L holds col N = L&15; contiguous K [k0+g*16, +16) of row N of the K-major source.
__device__ __forceinline__ v16bf load_b_frag(const u16* Bm, int ldb, int col, int k0, int lane) {
  int g = lane >> 4;
  return *reinterpret_cast<const v16bf*>(Bm + (size_t)col * ldb + k0 + g * 16);
}

// ---------------- scalar dense: Y = act((X@W + b) * BN); W row-major [K,N] ----------------
__global__ void dense_scalar(const float* __restrict__ X, const float* __restrict__ W,
                             const float* __restrict__ bias, float* __restrict__ Y,
                             int rows, int K, int N, int act) {
  int i = blockIdx.x * blockDim.x + threadIdx.x;
  if (i >= rows * N) return;
  int r = i / N, n = i % N;
  const float* x = X + (size_t)r * K;
  float s = 0.f;
  for (int k = 0; k < K; ++k) s += x[k] * W[(size_t)k * N + n];
  s += bias[n];
  if (act) { s *= BNC; s = s > 0.f ? s : 0.f; }
  Y[i] = s;
}

// ---------------- build padded node tensor [B, NP, D] (pads zeroed every call) ----------------
__global__ void build_node(const float* __restrict__ Xpep, const float* __restrict__ Xpro,
                           const float* __restrict__ pepres, const float* __restrict__ prores,
                           const float* __restrict__ nodeemb, float* __restrict__ node) {
  size_t i = (size_t)blockIdx.x * blockDim.x + threadIdx.x;
  if (i >= (size_t)B_ * NP * D_) return;
  int d = (int)(i % D_);
  size_t t = i / D_;
  int n = (int)(t % NP);
  int b = (int)(t / NP);
  float v;
  if      (n == 0)   v = Xpep[(size_t)b * D_ + d];
  else if (n == 1)   v = Xpro[(size_t)b * D_ + d];
  else if (n < 52)   v = pepres[((size_t)b * NPEPC + (n - 2)) * D_ + d];
  else if (n < 728)  v = prores[((size_t)b * NPROC + (n - 52)) * D_ + d];
  else if (n < NV)   v = nodeemb[(size_t)(n - 728) * D_ + d];
  else               v = 0.f;
  node[i] = v;
}

// ---------------- per-row norm + bf16 row-major & transposed copies ----------------
__global__ void norm_convert(const float* __restrict__ node, float* __restrict__ dist,
                             u16* __restrict__ xb, u16* __restrict__ xT) {
  int bn = blockIdx.x;               // B_*NP blocks
  int b = bn / NP, n = bn % NP;
  const float* row = node + (size_t)bn * D_;
  __shared__ float red[256];
  float s = 0.f;
  for (int d = threadIdx.x; d < D_; d += 256) { float v = row[d]; s += v * v; }
  red[threadIdx.x] = s; __syncthreads();
  for (int o = 128; o > 0; o >>= 1) { if (threadIdx.x < o) red[threadIdx.x] += red[threadIdx.x + o]; __syncthreads(); }
  if (threadIdx.x == 0) dist[bn] = sqrtf(red[0] + 1e-8f);
  for (int d = threadIdx.x; d < D_; d += 256) {
    u16 h = f2bf(row[d]);
    xb[(size_t)bn * D_ + d] = h;
    xT[((size_t)b * D_ + d) * NP + n] = h;
  }
}

// ---------------- WMMA: cos = relu(node.node^T / (|n||m|+eps)), diag=1 ----------------
// One wave computes a 32x64 tile: 2 A frags x 4 B frags -> 8 WMMA per k-step.
__global__ void __launch_bounds__(128, 1)
gemm_cos(const u16* __restrict__ node, const float* __restrict__ dist,
         u16* __restrict__ cosb) {
  int wave = threadIdx.x >> 5, lane = threadIdx.x & 31;
  long gtile = (long)blockIdx.x * 4 + wave;
  const long total = (long)B_ * MB_NP * NB_NP;
  if (gtile >= total) return;
  int b   = (int)(gtile / (MB_NP * NB_NP));
  int t   = (int)(gtile % (MB_NP * NB_NP));
  int tmb = t / NB_NP, tnb = t % NB_NP;
  const u16* Nb = node + (size_t)b * NP * D_;
  int ar0 = tmb * 32 + (lane & 15);
  int cb  = tnb * 64 + (lane & 15);
  v8f c[2][4] = {{{}, {}, {}, {}}, {{}, {}, {}, {}}};
#pragma unroll 2
  for (int k0 = 0; k0 < D_; k0 += 32) {
    v16bf a0 = load_a_frag(Nb, D_, ar0,      k0, lane);
    v16bf a1 = load_a_frag(Nb, D_, ar0 + 16, k0, lane);
#pragma unroll
    for (int j = 0; j < 4; ++j) {
      v16bf bb = load_b_frag(Nb, D_, cb + j * 16, k0, lane);
      c[0][j] = __builtin_amdgcn_wmma_f32_16x16x32_bf16(false, a0, false, bb, (short)0, c[0][j], false, false);
      c[1][j] = __builtin_amdgcn_wmma_f32_16x16x32_bf16(false, a1, false, bb, (short)0, c[1][j], false, false);
    }
  }
  const float* db = dist + (size_t)b * NP;
  int g = lane >> 4;
  u16* out = cosb + (size_t)b * NP * NP;
#pragma unroll
  for (int s = 0; s < 2; ++s) {
#pragma unroll
    for (int j = 0; j < 4; ++j) {
      int col = cb + j * 16;
      float dc = db[col];
#pragma unroll
      for (int e = 0; e < 8; ++e) {
        int r = tmb * 32 + s * 16 + 8 * g + e;
        float v = c[s][j][e] / (db[r] * dc + 1e-8f);
        if (r == col && r < NV) v = 1.0f;
        v = v > 0.f ? v : 0.f;
        out[(size_t)r * NP + col] = f2bf(v);
      }
    }
  }
}

// ---------------- row sums -> dinv (pads -> 0) ----------------
__global__ void rowsum_dinv(const u16* __restrict__ cosb, float* __restrict__ dinv) {
  int bn = blockIdx.x;               // B_*NP
  const u16* row = cosb + (size_t)bn * NP;
  __shared__ float red[256];
  float s = 0.f;
  for (int m = threadIdx.x; m < NP; m += 256) s += bf2f(row[m]);
  red[threadIdx.x] = s; __syncthreads();
  for (int o = 128; o > 0; o >>= 1) { if (threadIdx.x < o) red[threadIdx.x] += red[threadIdx.x + o]; __syncthreads(); }
  if (threadIdx.x == 0) { float v = red[0]; dinv[bn] = v > 1e-12f ? rsqrtf(v) : 0.f; }
}

// ---------------- pL = dinv_n * cos * dinv_m (in place, bf16) ----------------
__global__ void scale_pl(u16* __restrict__ cosb, const float* __restrict__ dinv) {
  size_t i = (size_t)blockIdx.x * blockDim.x + threadIdx.x;
  if (i >= (size_t)B_ * NP * NP) return;
  int m = (int)(i % NP);
  size_t t = i / NP;
  int n = (int)(t % NP);
  int b = (int)(t / NP);
  float v = bf2f(cosb[i]) * dinv[(size_t)b * NP + n] * dinv[(size_t)b * NP + m];
  cosb[i] = f2bf(v);
}

// ---------------- WMMA: y = pL @ x  (B-src = xT, ld=NP); 32x64 tiles ----------------
__global__ void __launch_bounds__(128, 1)
gemm_pl_x(const u16* __restrict__ pl, const u16* __restrict__ xT,
          u16* __restrict__ y) {
  int wave = threadIdx.x >> 5, lane = threadIdx.x & 31;
  long gtile = (long)blockIdx.x * 4 + wave;
  const long total = (long)B_ * MB_NP * NB_D;
  if (gtile >= total) return;
  int b   = (int)(gtile / (MB_NP * NB_D));
  int t   = (int)(gtile % (MB_NP * NB_D));
  int tmb = t / NB_D, tnb = t % NB_D;
  const u16* Ab = pl + (size_t)b * NP * NP;
  const u16* Bb = xT + (size_t)b * D_ * NP;
  int ar0 = tmb * 32 + (lane & 15);
  int cb  = tnb * 64 + (lane & 15);
  v8f c[2][4] = {{{}, {}, {}, {}}, {{}, {}, {}, {}}};
#pragma unroll 2
  for (int k0 = 0; k0 < NP; k0 += 32) {
    v16bf a0 = load_a_frag(Ab, NP, ar0,      k0, lane);
    v16bf a1 = load_a_frag(Ab, NP, ar0 + 16, k0, lane);
#pragma unroll
    for (int j = 0; j < 4; ++j) {
      v16bf bb = load_b_frag(Bb, NP, cb + j * 16, k0, lane);
      c[0][j] = __builtin_amdgcn_wmma_f32_16x16x32_bf16(false, a0, false, bb, (short)0, c[0][j], false, false);
      c[1][j] = __builtin_amdgcn_wmma_f32_16x16x32_bf16(false, a1, false, bb, (short)0, c[1][j], false, false);
    }
  }
  int g = lane >> 4;
  u16* yo = y + (size_t)b * NP * D_;
#pragma unroll
  for (int s = 0; s < 2; ++s) {
#pragma unroll
    for (int j = 0; j < 4; ++j) {
      int col = cb + j * 16;
#pragma unroll
      for (int e = 0; e < 8; ++e) {
        int r = tmb * 32 + s * 16 + 8 * g + e;
        yo[(size_t)r * D_ + col] = f2bf(c[s][j][e]);
      }
    }
  }
}

// ---------------- WMMA: x = relu((y@W + b)*BN) + x; writes xf, xb, xT; 32x64 tiles ----------------
__global__ void __launch_bounds__(128, 1)
gemm_layer(const u16* __restrict__ y, const u16* __restrict__ Wt,
           const float* __restrict__ bias, float* __restrict__ xf,
           u16* __restrict__ xb, u16* __restrict__ xT) {
  int wave = threadIdx.x >> 5, lane = threadIdx.x & 31;
  long gtile = (long)blockIdx.x * 4 + wave;
  const long total = (long)B_ * MB_NP * NB_D;
  if (gtile >= total) return;
  int b   = (int)(gtile / (MB_NP * NB_D));
  int t   = (int)(gtile % (MB_NP * NB_D));
  int tmb = t / NB_D, tnb = t % NB_D;
  const u16* Ab = y + (size_t)b * NP * D_;
  int ar0 = tmb * 32 + (lane & 15);
  int cb  = tnb * 64 + (lane & 15);
  v8f c[2][4] = {{{}, {}, {}, {}}, {{}, {}, {}, {}}};
#pragma unroll 2
  for (int k0 = 0; k0 < D_; k0 += 32) {
    v16bf a0 = load_a_frag(Ab, D_, ar0,      k0, lane);
    v16bf a1 = load_a_frag(Ab, D_, ar0 + 16, k0, lane);
#pragma unroll
    for (int j = 0; j < 4; ++j) {
      v16bf bb = load_b_frag(Wt, D_, cb + j * 16, k0, lane);
      c[0][j] = __builtin_amdgcn_wmma_f32_16x16x32_bf16(false, a0, false, bb, (short)0, c[0][j], false, false);
      c[1][j] = __builtin_amdgcn_wmma_f32_16x16x32_bf16(false, a1, false, bb, (short)0, c[1][j], false, false);
    }
  }
  int g = lane >> 4;
#pragma unroll
  for (int s = 0; s < 2; ++s) {
#pragma unroll
    for (int j = 0; j < 4; ++j) {
      int col = cb + j * 16;
      float bc = bias[col];
#pragma unroll
      for (int e = 0; e < 8; ++e) {
        int r = tmb * 32 + s * 16 + 8 * g + e;
        if (r < NV) {
          float v = (c[s][j][e] + bc) * BNC;
          v = v > 0.f ? v : 0.f;
          size_t idx = ((size_t)b * NP + r) * D_ + col;
          float nv = v + xf[idx];
          xf[idx] = nv;
          u16 h = f2bf(nv);
          xb[idx] = h;
          xT[((size_t)b * D_ + col) * NP + r] = h;
        }
      }
    }
  }
}

// ---------------- weight transpose/convert: Wt[n*K+k] = bf16(W[k*N+n]) ----------------
__global__ void wt_bf16(const float* __restrict__ W, u16* __restrict__ Wt, int K, int N) {
  size_t i = (size_t)blockIdx.x * blockDim.x + threadIdx.x;
  if (i >= (size_t)K * N) return;
  int k = (int)(i / N), n = (int)(i % N);
  Wt[(size_t)n * K + k] = f2bf(W[i]);
}

// ---------------- residue rows: R = x[res]*x[anchor] (bf16) ----------------
__global__ void build_R(const float* __restrict__ xf, u16* __restrict__ R) {
  size_t i = (size_t)blockIdx.x * blockDim.x + threadIdx.x;
  if (i >= (size_t)ROWS_TOT * D_) return;
  int d = (int)(i % D_);
  size_t r = i / D_;
  int b, nidx, mrow;
  if (r < ROWS_PEP) { b = (int)(r / NPEPC); nidx = 2  + (int)(r % NPEPC); mrow = 1; }
  else { size_t rr = r - ROWS_PEP; b = (int)(rr / NPROC); nidx = 52 + (int)(rr % NPROC); mrow = 0; }
  const float* xb_ = xf + (size_t)b * NP * D_;
  R[i] = f2bf(xb_[(size_t)nidx * D_ + d] * xb_[(size_t)mrow * D_ + d]);
}

// ---------------- WMMA heads: out = relu((R@W1+b1)*BN) @ w2 + b2 (fused, 32-row tiles) ----------------
__global__ void __launch_bounds__(128, 1)
gemm_heads(const u16* __restrict__ R,
           const u16* __restrict__ pW1t, const float* __restrict__ pb1,
           const float* __restrict__ pw2, const float* __restrict__ pb2,
           const u16* __restrict__ qW1t, const float* __restrict__ qb1,
           const float* __restrict__ qw2, const float* __restrict__ qb2,
           float* __restrict__ out) {
  int wave = threadIdx.x >> 5, lane = threadIdx.x & 31;
  int tile = blockIdx.x * 4 + wave;
  if (tile >= ROWS_TOT / 32) return;
  int row0 = tile * 32;
  bool isPep = row0 < ROWS_PEP;          // 3200 % 32 == 0: tiles never straddle pep/pro
  const u16*  Wt = isPep ? pW1t : qW1t;
  const float* b1 = isPep ? pb1 : qb1;
  const float* w2 = isPep ? pw2 : qw2;
  float b2 = isPep ? pb2[0] : qb2[0];
  int ar0 = row0 + (lane & 15);
  float acc[2][8] = {{0.f,0.f,0.f,0.f,0.f,0.f,0.f,0.f},
                     {0.f,0.f,0.f,0.f,0.f,0.f,0.f,0.f}};
#pragma unroll 1
  for (int ng = 0; ng < H_ / 64; ++ng) {   // 16 groups of 4 column tiles
    int cb = ng * 64 + (lane & 15);
    v8f c[2][4] = {{{}, {}, {}, {}}, {{}, {}, {}, {}}};
#pragma unroll 2
    for (int k0 = 0; k0 < D_; k0 += 32) {
      v16bf a0 = load_a_frag(R, D_, ar0,      k0, lane);
      v16bf a1 = load_a_frag(R, D_, ar0 + 16, k0, lane);
#pragma unroll
      for (int j = 0; j < 4; ++j) {
        v16bf bb = load_b_frag(Wt, D_, cb + j * 16, k0, lane);
        c[0][j] = __builtin_amdgcn_wmma_f32_16x16x32_bf16(false, a0, false, bb, (short)0, c[0][j], false, false);
        c[1][j] = __builtin_amdgcn_wmma_f32_16x16x32_bf16(false, a1, false, bb, (short)0, c[1][j], false, false);
      }
    }
#pragma unroll
    for (int j = 0; j < 4; ++j) {
      int col = cb + j * 16;
      float w2c = w2[col], bc = b1[col];
#pragma unroll
      for (int s = 0; s < 2; ++s) {
#pragma unroll
        for (int e = 0; e < 8; ++e) {
          float v = (c[s][j][e] + bc) * BNC;
          v = v > 0.f ? v : 0.f;
          acc[s][e] += v * w2c;
        }
      }
    }
  }
  // reduce over the 16 columns held by lanes sharing a half-wave
#pragma unroll
  for (int off = 1; off < 16; off <<= 1)
#pragma unroll
    for (int s = 0; s < 2; ++s)
#pragma unroll
      for (int e = 0; e < 8; ++e) acc[s][e] += __shfl_xor(acc[s][e], off, 32);
  if ((lane & 15) == 0) {
    int g = lane >> 4;
#pragma unroll
    for (int s = 0; s < 2; ++s)
#pragma unroll
      for (int e = 0; e < 8; ++e) {
        int r = row0 + s * 16 + 8 * g + e;
        out[64 + r] = acc[s][e] + b2;  // pep rows then pro rows are contiguous in d_out
      }
  }
}

// ---------------- interaction head pieces ----------------
__global__ void node_embed_k(const float* __restrict__ xf, float* __restrict__ ne) {
  int i = blockIdx.x * blockDim.x + threadIdx.x;
  if (i >= B_ * D_) return;
  int b = i / D_, d = i % D_;
  const float* xb_ = xf + (size_t)b * NP * D_;
  ne[i] = xb_[d] * xb_[(size_t)D_ + d];
}

__global__ void int_head(const float* __restrict__ h2, const float* __restrict__ lW3,
                         const float* __restrict__ lb3, float* __restrict__ out) {
  int b = blockIdx.x, t = threadIdx.x;
  __shared__ float red[256];
  float s = 0.f;
  for (int k = t; k < D_; k += 256) s += h2[(size_t)b * D_ + k] * lW3[k];
  red[t] = s; __syncthreads();
  for (int o = 128; o > 0; o >>= 1) { if (t < o) red[t] += red[t + o]; __syncthreads(); }
  if (t == 0) { float v = red[0] + lb3[0]; out[b] = 1.f / (1.f + expf(-v)); }
}

// ================================================================================
extern "C" void kernel_launch(void* const* d_in, const int* in_sizes, int n_in,
                              void* d_out, int out_size, void* d_ws, size_t ws_size,
                              hipStream_t stream) {
  const float* pep_feature = (const float*)d_in[0];
  const float* pro_feature = (const float*)d_in[1];
  const float* pep_res     = (const float*)d_in[2];
  const float* pro_res     = (const float*)d_in[3];
  const float* node_emb    = (const float*)d_in[4];
  const float* pepW1 = (const float*)d_in[5];  const float* pepb1 = (const float*)d_in[6];
  const float* pepW2 = (const float*)d_in[7];  const float* pepb2 = (const float*)d_in[8];
  const float* proW1 = (const float*)d_in[9];  const float* prob1 = (const float*)d_in[10];
  const float* proW2 = (const float*)d_in[11]; const float* prob2 = (const float*)d_in[12];
  const float* gW[3] = {(const float*)d_in[13], (const float*)d_in[15], (const float*)d_in[17]};
  const float* gb[3] = {(const float*)d_in[14], (const float*)d_in[16], (const float*)d_in[18]};
  const float* pbsW1 = (const float*)d_in[19]; const float* pbsb1 = (const float*)d_in[20];
  const float* pbsW2 = (const float*)d_in[21]; const float* pbsb2 = (const float*)d_in[22];
  const float* qbsW1 = (const float*)d_in[23]; const float* qbsb1 = (const float*)d_in[24];
  const float* qbsW2 = (const float*)d_in[25]; const float* qbsb2 = (const float*)d_in[26];
  const float* lW1 = (const float*)d_in[27]; const float* lb1 = (const float*)d_in[28];
  const float* lW2 = (const float*)d_in[29]; const float* lb2 = (const float*)d_in[30];
  const float* lW3 = (const float*)d_in[31]; const float* lb3 = (const float*)d_in[32];
  float* out = (float*)d_out;

  // ----- workspace carve (256B aligned) -----
  size_t off = 0;
  auto carve = [&](size_t bytes) -> void* {
    void* p = (char*)d_ws + off;
    off += (bytes + 255) & ~(size_t)255;
    return p;
  };
  float* hidP  = (float*)carve((size_t)B_ * H_ * 4);
  float* hidQ  = (float*)carve((size_t)B_ * H_ * 4);
  float* Xpep  = (float*)carve((size_t)B_ * D_ * 4);
  float* Xpro  = (float*)carve((size_t)B_ * D_ * 4);
  float* nodef = (float*)carve((size_t)B_ * NP * D_ * 4);   // x (f32, residual)
  float* dist  = (float*)carve((size_t)B_ * NP * 4);
  float* dinv  = (float*)carve((size_t)B_ * NP * 4);
  u16*   xb    = (u16*)  carve((size_t)B_ * NP * D_ * 2);   // x (bf16, row-major)
  u16*   xT    = (u16*)  carve((size_t)B_ * D_ * NP * 2);   // x^T (bf16)
  u16*   cosb  = (u16*)  carve((size_t)B_ * NP * NP * 2);   // cos -> pL (bf16)
  u16*   ybuf  = (u16*)  carve((size_t)B_ * NP * D_ * 2);   // pL@x (bf16)
  u16*   gWt0  = (u16*)  carve((size_t)D_ * D_ * 2);
  u16*   gWt1  = (u16*)  carve((size_t)D_ * D_ * 2);
  u16*   gWt2  = (u16*)  carve((size_t)D_ * D_ * 2);
  u16*   pW1t  = (u16*)  carve((size_t)D_ * H_ * 2);
  u16*   qW1t  = (u16*)  carve((size_t)D_ * H_ * 2);
  u16*   Rbuf  = (u16*)  carve((size_t)ROWS_TOT * D_ * 2);
  float* ne    = (float*)carve((size_t)B_ * D_ * 4);
  float* eh1   = (float*)carve((size_t)B_ * D_ * 4);
  float* eh2   = (float*)carve((size_t)B_ * D_ * 4);
  u16*   gWt[3] = {gWt0, gWt1, gWt2};
  (void)in_sizes; (void)n_in; (void)out_size; (void)ws_size;

  // ----- 1) global-feature MLPs (tiny, scalar) -----
  dense_scalar<<<(B_ * H_ + 255) / 256, 256, 0, stream>>>(pep_feature, pepW1, pepb1, hidP, B_, D_, H_, 1);
  dense_scalar<<<(B_ * D_ + 255) / 256, 256, 0, stream>>>(hidP, pepW2, pepb2, Xpep, B_, H_, D_, 1);
  dense_scalar<<<(B_ * H_ + 255) / 256, 256, 0, stream>>>(pro_feature, proW1, prob1, hidQ, B_, D_, H_, 1);
  dense_scalar<<<(B_ * D_ + 255) / 256, 256, 0, stream>>>(hidQ, proW2, prob2, Xpro, B_, H_, D_, 1);

  // ----- 2) node tensor + bf16 copies + norms -----
  build_node<<<(unsigned)(((size_t)B_ * NP * D_ + 255) / 256), 256, 0, stream>>>(
      Xpep, Xpro, pep_res, pro_res, node_emb, nodef);
  norm_convert<<<B_ * NP, 256, 0, stream>>>(nodef, dist, xb, xT);

  // ----- 3) bf16 transposed weights -----
  for (int l = 0; l < 3; ++l)
    wt_bf16<<<(D_ * D_ + 255) / 256, 256, 0, stream>>>(gW[l], gWt[l], D_, D_);
  wt_bf16<<<(D_ * H_ + 255) / 256, 256, 0, stream>>>(pbsW1, pW1t, D_, H_);
  wt_bf16<<<(D_ * H_ + 255) / 256, 256, 0, stream>>>(qbsW1, qW1t, D_, H_);

  // ----- 4) adjacency: WMMA cos, row sums, pL scaling -----
  gemm_cos<<<(B_ * MB_NP * NB_NP) / 4, 128, 0, stream>>>(xb, dist, cosb);
  rowsum_dinv<<<B_ * NP, 256, 0, stream>>>(cosb, dinv);
  scale_pl<<<(unsigned)(((size_t)B_ * NP * NP + 255) / 256), 256, 0, stream>>>(cosb, dinv);

  // ----- 5) 3 GCN layers (two WMMA GEMMs each, fused residual epilogue) -----
  for (int l = 0; l < 3; ++l) {
    gemm_pl_x <<<(B_ * MB_NP * NB_D) / 4, 128, 0, stream>>>(cosb, xT, ybuf);
    gemm_layer<<<(B_ * MB_NP * NB_D) / 4, 128, 0, stream>>>(ybuf, gWt[l], gb[l], nodef, xb, xT);
  }

  // ----- 6) residue heads (fused 2-layer WMMA MLP) -----
  build_R<<<(unsigned)(((size_t)ROWS_TOT * D_ + 255) / 256), 256, 0, stream>>>(nodef, Rbuf);
  gemm_heads<<<(ROWS_TOT / 32 + 3) / 4, 128, 0, stream>>>(
      Rbuf, pW1t, pbsb1, pbsW2, pbsb2, qW1t, qbsb1, qbsW2, qbsb2, out);

  // ----- 7) interaction head -----
  node_embed_k<<<(B_ * D_ + 255) / 256, 256, 0, stream>>>(nodef, ne);
  dense_scalar<<<(B_ * D_ + 255) / 256, 256, 0, stream>>>(ne,  lW1, lb1, eh1, B_, D_, D_, 1);
  dense_scalar<<<(B_ * D_ + 255) / 256, 256, 0, stream>>>(eh1, lW2, lb2, eh2, B_, D_, D_, 1);
  int_head<<<B_, 256, 0, stream>>>(eh2, lW3, lb3, out);
}
